// MoEFSL_38019050504976
// MI455X (gfx1250) — matrix operand order repaired and checked by reading, not compiled, not run on hardware
//
#include <hip/hip_runtime.h>

// ---------------------------------------------------------------------------
// Problem constants (from reference): D=384, E=16, H=16, C=64, N=262144
// ---------------------------------------------------------------------------
#define DD  384
#define EE  16
#define HH  16
#define CC  64

typedef __attribute__((ext_vector_type(16))) __bf16        v16bf;
typedef __attribute__((ext_vector_type(8)))  float         v8f;
typedef __attribute__((ext_vector_type(4)))  unsigned int  v4u;
typedef __attribute__((ext_vector_type(4)))  int           v4i;
typedef __attribute__((ext_vector_type(8)))  int           v8i;

union Fr16 { v16bf v; unsigned int w[8]; unsigned short u[16]; };
union Acc8 { v8f   v; float f[8]; };

// native f32 -> bf16 (RNE); backend picks the hardware convert on gfx1250
__device__ __forceinline__ unsigned short f2bf(float f) {
    return __builtin_bit_cast(unsigned short, static_cast<__bf16>(f));
}

__device__ __forceinline__ float gelu_exact(float v) {
    return 0.5f * v * (1.0f + erff(v * 0.70710678118654752f));
}

__device__ __forceinline__ v8f wmma_bf16(v16bf a, v16bf b, v8f c) {
    return __builtin_amdgcn_wmma_f32_16x16x32_bf16(
        false, a, false, b, (short)0, c, false, false);
}

// ---------------------------------------------------------------------------
// Weight packing into BF16 WMMA B-fragment layout:
//   fragment index = ((kc * nTiles + nt) * 32 + lane) * 16 + i
//   lane = (n % 16) + 16 * (((k % 32) / 8) % 2),  i = (k % 8) + 8 * ((k % 32) / 16)
// ---------------------------------------------------------------------------
__global__ void pack_w1(const float* __restrict__ W1, unsigned short* __restrict__ W1p) {
    int g = blockIdx.x * blockDim.x + threadIdx.x;
    if (g >= EE * DD * HH) return;
    int e = g / (DD * HH);
    int r = g % (DD * HH);
    int d = r / HH;           // K (0..383)
    int h = r % HH;
    int c = e * HH + h;       // N (0..255)
    int nt = c >> 4, nn = c & 15;
    int kc = d >> 5, kl = d & 31;
    W1p[(((kc * 16 + nt) * 32) + nn + 16 * ((kl >> 3) & 1)) * 16 + ((kl & 7) + ((kl >> 4) << 3))] = f2bf(W1[g]);
}

__global__ void pack_w2(const float* __restrict__ W2, unsigned short* __restrict__ W2p) {
    int g = blockIdx.x * blockDim.x + threadIdx.x;
    if (g >= EE * HH * DD) return;
    int e = g / (HH * DD);
    int h = (g / DD) % HH;
    int d = g % DD;           // N (0..383)
    int kH = e * HH + h;      // K (0..255)
    int kc = kH >> 5, kl = kH & 31;
    int dt = d >> 4, nn = d & 15;
    W2p[(((kc * 24 + dt) * 32) + nn + 16 * ((kl >> 3) & 1)) * 16 + ((kl & 7) + ((kl >> 4) << 3))] = f2bf(W2[g]);
}

__global__ void pack_wh(const float* __restrict__ Wh, unsigned short* __restrict__ Whp) {
    int g = blockIdx.x * blockDim.x + threadIdx.x;
    if (g >= DD * CC) return;
    int d = g / CC;           // K (0..383)
    int c = g % CC;           // N (0..63)
    int kc = d >> 5, kl = d & 31;
    int ct = c >> 4, nn = c & 15;
    Whp[(((kc * 4 + ct) * 32) + nn + 16 * ((kl >> 3) & 1)) * 16 + ((kl & 7) + ((kl >> 4) << 3))] = f2bf(Wh[g]);
}

__global__ void init_accum(float* accum) {
    if (threadIdx.x < 4) accum[threadIdx.x] = 0.0f;
}

// ---------------------------------------------------------------------------
// Shuffle-based softmax stats on WMMA C-layout accumulators.
// r[4] = 4 N-tiles of logits (64 cols) for 8 rows (this lane's M-half).
// Cross-lane reductions over the 16-lane group holding n = 0..15.
// ---------------------------------------------------------------------------
__device__ __forceinline__ void softstats(const Acc8* r, const int* lblj, int lane,
                                          float& ceAcc, float& entAcc) {
    const int halfsel = lane & 16;
    #pragma unroll
    for (int j = 0; j < 8; ++j) {
        float v0 = r[0].f[j], v1 = r[1].f[j], v2 = r[2].f[j], v3 = r[3].f[j];
        float mx = fmaxf(fmaxf(v0, v1), fmaxf(v2, v3));
        mx = fmaxf(mx, __shfl_xor(mx, 1));
        mx = fmaxf(mx, __shfl_xor(mx, 2));
        mx = fmaxf(mx, __shfl_xor(mx, 4));
        mx = fmaxf(mx, __shfl_xor(mx, 8));
        float e0 = __expf(v0 - mx), e1 = __expf(v1 - mx);
        float e2 = __expf(v2 - mx), e3 = __expf(v3 - mx);
        float s1 = (e0 + e1) + (e2 + e3);
        float s2 = (e0 * v0 + e1 * v1) + (e2 * v2 + e3 * v3);
        s1 += __shfl_xor(s1, 1);  s2 += __shfl_xor(s2, 1);
        s1 += __shfl_xor(s1, 2);  s2 += __shfl_xor(s2, 2);
        s1 += __shfl_xor(s1, 4);  s2 += __shfl_xor(s2, 4);
        s1 += __shfl_xor(s1, 8);  s2 += __shfl_xor(s2, 8);
        float lse = mx + __logf(s1);
        int lab = lblj[j];
        float cand = (lab < 16) ? v0 : (lab < 32) ? v1 : (lab < 48) ? v2 : v3;
        float lval = __shfl(cand, halfsel + (lab & 15));
        ceAcc  += lse - lval;
        entAcc += lse - s2 / s1;
    }
}

// ---------------------------------------------------------------------------
// Main fused kernel: one wave (32 lanes) per 16-token tile.
// ---------------------------------------------------------------------------
__global__ __launch_bounds__(32) void moe_loss_main(
    const float* __restrict__ x, const int* __restrict__ labels,
    const float* __restrict__ b1, const float* __restrict__ b2,
    const float* __restrict__ bh,
    const unsigned short* __restrict__ W1p,
    const unsigned short* __restrict__ W2p,
    const unsigned short* __restrict__ Whp,
    float* __restrict__ accum)
{
    const int lane = threadIdx.x;
    const int tile = blockIdx.x;

    // 24 KB pool: xstage (16x384 f32) early; fragH (8K) + fragO (12K) later.
    __shared__ __attribute__((aligned(32))) unsigned char pool[16 * DD * 4];
    __shared__ int lbls[16];

    float* xstage = (float*)pool;
    unsigned short (*fragH)[32][16] = (unsigned short (*)[32][16])(pool);         // [8][32][16]
    unsigned short (*fragO)[32][16] = (unsigned short (*)[32][16])(pool + 8192);  // [12][32][16]

    if (lane < 16) lbls[lane] = labels[tile * 16 + lane];

    const int mrow = lane & 15;   // A-frag row owned by this lane
    const int half = lane >> 4;   // K-half (A) / M-half (C)
    const int ncol = lane & 15;   // C-layout column

    // ---- Stage x tile: TDM async tensor load when available ---------------
#if __has_builtin(__builtin_amdgcn_tensor_load_to_lds) && __has_builtin(__builtin_amdgcn_s_wait_tensorcnt)
    {
        unsigned long long gaddr =
            (unsigned long long)(uintptr_t)x + (unsigned long long)tile * (16ull * DD * 4ull);
        unsigned int ldsoff = (unsigned int)(uintptr_t)pool;
        v4u g0;
        g0[0] = 1u;                                                  // count=1 user descriptor
        g0[1] = ldsoff;                                              // lds_addr
        g0[2] = (unsigned int)gaddr;                                 // global_addr[31:0]
        g0[3] = (unsigned int)((gaddr >> 32) & 0x01FFFFFFull)        // global_addr[56:32]
              | (2u << 30);                                          // type=2 (image)
        v8i g1;
        g1[0] = (int)(2u << 16);                   // data_size = 4 bytes
        g1[1] = (int)(((unsigned)DD) << 16);       // tensor_dim0[15:0] = 384
        g1[2] = (int)(16u << 16);                  // tensor_dim1[15:0] = 16
        g1[3] = (int)(((unsigned)DD) << 16);       // tile_dim0 = 384
        g1[4] = 16;                                // tile_dim1 = 16
        g1[5] = DD;                                // tensor_dim0_stride = 384
        g1[6] = 0;
        g1[7] = 0;
        v4i gz = {0, 0, 0, 0};
    #if __clang_major__ >= 23
        v8i gz8 = {0, 0, 0, 0, 0, 0, 0, 0};
        __builtin_amdgcn_tensor_load_to_lds(g0, g1, gz, gz, gz8, 0);
    #else
        __builtin_amdgcn_tensor_load_to_lds(g0, g1, gz, gz, 0);
    #endif
        __builtin_amdgcn_s_wait_tensorcnt(0);
    }
    __syncthreads();
    const float* xrow = xstage + mrow * DD;
#else
    __syncthreads();
    const float* xrow = x + (size_t)(tile * 16 + mrow) * DD;
#endif

    // ---- Build X A-fragments (native f32 -> bf16 casts) -------------------
    v16bf aX[12];
    #pragma unroll
    for (int kc = 0; kc < 12; ++kc) {
        int c0 = kc * 32 + half * 8;        // K 0..7  (lane<16) / 8..15  (lane>=16)
        int c1 = kc * 32 + 16 + half * 8;   // K 16..23           / 24..31
        float4 p0 = *(const float4*)(xrow + c0);
        float4 p1 = *(const float4*)(xrow + c0 + 4);
        float4 p2 = *(const float4*)(xrow + c1);
        float4 p3 = *(const float4*)(xrow + c1 + 4);
        v16bf av;
        av[0]  = (__bf16)p0.x; av[1]  = (__bf16)p0.y; av[2]  = (__bf16)p0.z; av[3]  = (__bf16)p0.w;
        av[4]  = (__bf16)p1.x; av[5]  = (__bf16)p1.y; av[6]  = (__bf16)p1.z; av[7]  = (__bf16)p1.w;
        av[8]  = (__bf16)p2.x; av[9]  = (__bf16)p2.y; av[10] = (__bf16)p2.z; av[11] = (__bf16)p2.w;
        av[12] = (__bf16)p3.x; av[13] = (__bf16)p3.y; av[14] = (__bf16)p3.z; av[15] = (__bf16)p3.w;
        aX[kc] = av;
    }

    int lblj[8];
    #pragma unroll
    for (int j = 0; j < 8; ++j) lblj[j] = lbls[j + 8 * half];

    const v16bf* W1f = (const v16bf*)W1p;
    const v16bf* W2f = (const v16bf*)W2p;
    const v16bf* Whf = (const v16bf*)Whp;

    // ---- GEMM1: Hall = X @ W1f; fused bias + GELU + expert mask ----------
    // N-tile nt == expert index (H == 16); repack masked tile as A-frags.
    for (int nt = 0; nt < EE; ++nt) {
        v8f acc = {};
        #pragma unroll
        for (int kc = 0; kc < 12; ++kc)
            acc = wmma_bf16(aX[kc], W1f[(kc * 16 + nt) * 32 + lane], acc);

        int colH   = nt * 16 + ncol;          // K index of GEMM2 (0..255)
        float bias = b1[colH];
        int kc2    = colH >> 5;
        int kl     = colH & 31;
        int halfA  = (kl >> 3) & 1;
        int ii     = (kl & 7) + ((kl >> 4) << 3);
        Acc8 r; r.v = acc;
        #pragma unroll
        for (int j = 0; j < 8; ++j) {
            int mm = j + 8 * half;
            float vv = (lblj[j] == nt) ? gelu_exact(r.f[j] + bias) : 0.0f;
            fragH[kc2][mm + 16 * halfA][ii] = f2bf(vv);
        }
    }

    // ---- enc_logits = X @ Wh + bh  (kept in C-layout registers) ----------
    Acc8 rE[4];
    for (int ct = 0; ct < 4; ++ct) {
        v8f acc = {};
        #pragma unroll
        for (int kc = 0; kc < 12; ++kc)
            acc = wmma_bf16(aX[kc], Whf[(kc * 4 + ct) * 32 + lane], acc);
        float bhv = bh[ct * 16 + ncol];
        rE[ct].v = acc;
        #pragma unroll
        for (int j = 0; j < 8; ++j) rE[ct].f[j] += bhv;
    }
    __syncthreads();

    // ---- GEMM2: Out = Hmask @ W2f + b2[label]; repack as A-frags ---------
    v16bf aH[8];
    #pragma unroll
    for (int kc = 0; kc < 8; ++kc)
        aH[kc] = *(const v16bf*)&fragH[kc][lane][0];

    for (int dt = 0; dt < 24; ++dt) {
        v8f acc = {};
        #pragma unroll
        for (int kc = 0; kc < 8; ++kc)
            acc = wmma_bf16(aH[kc], W2f[(kc * 24 + dt) * 32 + lane], acc);

        int d     = dt * 16 + ncol;           // K index of GEMM4 (0..383)
        int kc4   = d >> 5;
        int kl    = d & 31;
        int halfA = (kl >> 3) & 1;
        int ii    = (kl & 7) + ((kl >> 4) << 3);
        Acc8 r; r.v = acc;
        #pragma unroll
        for (int j = 0; j < 8; ++j) {
            int mm = j + 8 * half;
            float vv = r.f[j] + b2[(size_t)lblj[j] * DD + d];
            fragO[kc4][mm + 16 * halfA][ii] = f2bf(vv);
        }
    }
    __syncthreads();

    // ---- mlp_logits = Out @ Wh + bh --------------------------------------
    v16bf aO[12];
    #pragma unroll
    for (int kc = 0; kc < 12; ++kc)
        aO[kc] = *(const v16bf*)&fragO[kc][lane][0];

    Acc8 rM[4];
    for (int ct = 0; ct < 4; ++ct) {
        v8f acc = {};
        #pragma unroll
        for (int kc = 0; kc < 12; ++kc)
            acc = wmma_bf16(aO[kc], Whf[(kc * 4 + ct) * 32 + lane], acc);
        float bhv = bh[ct * 16 + ncol];
        rM[ct].v = acc;
        #pragma unroll
        for (int j = 0; j < 8; ++j) rM[ct].f[j] += bhv;
    }

    // ---- CE + entropy via cross-lane shuffles ----------------------------
    float ceE = 0.f, entE = 0.f, ceM = 0.f, entM = 0.f;
    softstats(rE, lblj, lane, ceE, entE);
    softstats(rM, lblj, lane, ceM, entM);

    // fold the two M-halves (each 16-lane group holds identical sums)
    ceE  += __shfl_xor(ceE,  16);
    entE += __shfl_xor(entE, 16);
    ceM  += __shfl_xor(ceM,  16);
    entM += __shfl_xor(entM, 16);

    if (lane == 0) {
        atomicAdd(&accum[0], ceE);
        atomicAdd(&accum[1], entE);
        atomicAdd(&accum[2], ceM);
        atomicAdd(&accum[3], entM);
    }
}

__global__ void finalize_loss(const float* __restrict__ accum,
                              float* __restrict__ out, float invN) {
    float ceE  = accum[0] * invN;
    float entE = accum[1] * invN;
    float ceM  = accum[2] * invN;
    float entM = accum[3] * invN;
    out[0] = ceM + 0.5f * (entM - entE) + 0.5f * ceE;
}

// ---------------------------------------------------------------------------
extern "C" void kernel_launch(void* const* d_in, const int* in_sizes, int n_in,
                              void* d_out, int out_size, void* d_ws, size_t ws_size,
                              hipStream_t stream) {
    const float* x      = (const float*)d_in[0];
    const int*   labels = (const int*)d_in[1];
    const float* W1     = (const float*)d_in[2];
    const float* b1     = (const float*)d_in[3];
    const float* W2     = (const float*)d_in[4];
    const float* b2     = (const float*)d_in[5];
    const float* Wh     = (const float*)d_in[6];
    const float* bh     = (const float*)d_in[7];

    const int N = in_sizes[1];        // token count (262144)

    char* ws = (char*)d_ws;
    float*          accum = (float*)ws;                                  // 4 f32
    unsigned short* W1p   = (unsigned short*)(ws + 64);                  // 12*16*512 bf16
    unsigned short* W2p   = (unsigned short*)(ws + 64 + 196608);         //  8*24*512 bf16
    unsigned short* Whp   = (unsigned short*)(ws + 64 + 2 * 196608);     // 12* 4*512 bf16

    init_accum<<<1, 32, 0, stream>>>(accum);
    pack_w1<<<(EE * DD * HH + 255) / 256, 256, 0, stream>>>(W1, W1p);
    pack_w2<<<(EE * HH * DD + 255) / 256, 256, 0, stream>>>(W2, W2p);
    pack_wh<<<(DD * CC + 255) / 256, 256, 0, stream>>>(Wh, Whp);

    moe_loss_main<<<N / 16, 32, 0, stream>>>(x, labels, b1, b2, bh,
                                             W1p, W2p, Whp, accum);

    finalize_loss<<<1, 1, 0, stream>>>(accum, (float*)d_out, 1.0f / (float)N);
}
